// QLSTM_65481071395457
// MI455X (gfx1250) — matrix operand order repaired and checked by reading, not compiled, not run on hardware
//
#include <hip/hip_runtime.h>
#include <hip/hip_bf16.h>

// ---------------------------------------------------------------------------
// QLSTM for MI455X (gfx1250): bf16 WMMA GEMMs, x-part hoisted out of the
// recurrence, per-step h-part GEMM computing all 4 gates per wave.
// ---------------------------------------------------------------------------

#define T_LEN 512
#define B_SZ  256
#define D_SZ  256
#define H_SZ  256
#define FANIN 512           // D + H
#define TB    (T_LEN * B_SZ)  // 131072

typedef unsigned short ushort_t;
typedef __attribute__((ext_vector_type(16))) __bf16 v16bf;
typedef __attribute__((ext_vector_type(8)))  float  v8f;
typedef __attribute__((ext_vector_type(4)))  unsigned int u32x4;

union FragU { v16bf v; u32x4 u[2]; };

__device__ __forceinline__ ushort_t f2bf(float x) {
    unsigned int u = __float_as_uint(x);
    u += 0x7FFFu + ((u >> 16) & 1u);   // round-to-nearest-even
    return (ushort_t)(u >> 16);
}

__device__ __forceinline__ float sigmoidf_(float x) {
    return 1.0f / (1.0f + expf(-x));
}

// A-matrix 16x32 bf16 fragment (ISA 7.12.2): lane L -> M = L%16,
// kbase = (L<16)?0:8 ; VGPR0..3 = k[kbase..kbase+7], VGPR4..7 = k[kbase+16..kbase+23]
__device__ __forceinline__ v16bf load_a_frag(const ushort_t* __restrict__ A,
                                             int lda, int row0, int k0) {
    int lane = threadIdx.x & 31;
    int m  = lane & 15;
    int kb = k0 + ((lane >> 4) << 3);
    const ushort_t* p = A + (size_t)(row0 + m) * lda + kb;
    FragU f;
    f.u[0] = *(const u32x4*)(p);        // k = kb .. kb+7
    f.u[1] = *(const u32x4*)(p + 16);   // k = kb+16 .. kb+23
    return f.v;
}

// B-matrix 32x16 bf16 fragment, B[k][n] = W[n][k] (row-major W, ldw elems):
// lane L -> N = L%16, kbase = (L<16)?0:8 ; VGPR0..3 = k[kbase..kbase+7], VGPR4..7 = k[kbase+16..kbase+23]
// => two 8-bf16 runs from W row (col0 + n) at k0 + kbase and k0 + kbase + 16.
__device__ __forceinline__ v16bf load_b_frag(const ushort_t* __restrict__ W,
                                             int ldw, int col0, int k0) {
    int lane = threadIdx.x & 31;
    int n  = lane & 15;
    int kb = k0 + ((lane >> 4) << 3);
    const ushort_t* p = W + (size_t)(col0 + n) * ldw + kb;
    FragU f;
    f.u[0] = *(const u32x4*)(p);        // k = kb .. kb+7
    f.u[1] = *(const u32x4*)(p + 16);   // k = kb+16 .. kb+23
    return f.v;
}

__device__ __forceinline__ v8f wmma_bf16(v16bf a, v16bf b, v8f c) {
    return __builtin_amdgcn_wmma_f32_16x16x32_bf16(
        /*neg_a=*/false, a, /*neg_b=*/false, b,
        /*c_mod=*/(short)0, c, /*reuse_a=*/false, /*reuse_b=*/false);
}

// ---------------------------------------------------------------------------
// Conversions / init
// ---------------------------------------------------------------------------
__global__ void k_conv_weights(const float* __restrict__ w0, const float* __restrict__ w1,
                               const float* __restrict__ w2, const float* __restrict__ w3,
                               ushort_t* __restrict__ Wbf) {
    int idx = blockIdx.x * blockDim.x + threadIdx.x;
    const int per = H_SZ * FANIN;
    if (idx >= 4 * per) return;
    int g = idx / per, r = idx - g * per;
    const float* W = (g == 0) ? w0 : (g == 1) ? w1 : (g == 2) ? w2 : w3;
    Wbf[idx] = f2bf(W[r]);
}

__global__ void k_bias(const float* __restrict__ b0, const float* __restrict__ t0,
                       const float* __restrict__ b1, const float* __restrict__ t1,
                       const float* __restrict__ b2, const float* __restrict__ t2,
                       const float* __restrict__ b3, const float* __restrict__ t3,
                       float* __restrict__ bias) {
    int idx = blockIdx.x * blockDim.x + threadIdx.x;
    if (idx >= 4 * H_SZ) return;
    int g = idx >> 8, h = idx & (H_SZ - 1);
    float v;
    if      (g == 0) v = b0[h] + t0[h];
    else if (g == 1) v = b1[h] + t1[h];
    else if (g == 2) v = b2[h] + t2[h];
    else             v = b3[h] + t3[h];
    bias[idx] = v;
}

__global__ void k_conv_inputs(const float* __restrict__ x, ushort_t* __restrict__ xbf,
                              int n) {
    int idx = blockIdx.x * blockDim.x + threadIdx.x;
    if (idx < n) xbf[idx] = f2bf(x[idx]);
}

__global__ void k_init_state(ushort_t* __restrict__ h0, float* __restrict__ cx) {
    int idx = blockIdx.x * blockDim.x + threadIdx.x;
    if (idx < B_SZ * H_SZ) { h0[idx] = 0; cx[idx] = 0.0f; }
}

// ---------------------------------------------------------------------------
// Big parallel GEMM: gate_pre[g][t*B+b][h] = x[t,b,:] @ Wx_g[h,:] + bias_g[h]
// One wave per 16x16 output tile, K = D = 256 (8 WMMA k-steps).
// ---------------------------------------------------------------------------
__global__ __launch_bounds__(256)
void k_xgemm(const ushort_t* __restrict__ xbf, const ushort_t* __restrict__ Wbf,
             const float* __restrict__ bias, float* __restrict__ gate_pre) {
    int wid  = (blockIdx.x * blockDim.x + threadIdx.x) >> 5;
    int lane = threadIdx.x & 31;
    // tiles: 4 gates * (TB/16 = 8192 mtiles) * (H/16 = 16 ntiles)
    int g     = wid >> 17;
    int rem   = wid & 0x1FFFF;
    int mtile = rem >> 4;
    int ntile = rem & 15;
    int row0 = mtile << 4, col0 = ntile << 4;

    const ushort_t* Wg = Wbf + (size_t)g * H_SZ * FANIN;
    v8f acc = {0.f, 0.f, 0.f, 0.f, 0.f, 0.f, 0.f, 0.f};
    #pragma unroll
    for (int k0 = 0; k0 < D_SZ; k0 += 32) {
        v16bf a = load_a_frag(xbf, D_SZ, row0, k0);
        v16bf b = load_b_frag(Wg, FANIN, col0, k0);
        acc = wmma_bf16(a, b, acc);
    }
    int moff = (lane >> 4) << 3;
    int col  = col0 + (lane & 15);
    float bv = bias[g * H_SZ + col];
    float* gp = gate_pre + (size_t)g * TB * H_SZ;
    #pragma unroll
    for (int r = 0; r < 8; ++r)
        gp[(size_t)(row0 + moff + r) * H_SZ + col] = acc[r] + bv;
}

// ---------------------------------------------------------------------------
// Recurrent step: one wave per 16x16 (batch x hidden) tile, all 4 gates.
// acc initialized from gate_pre (x-part + bias), accumulate h @ Wh.T, then
// LSTM cell elementwise. h double-buffered across launches (bf16).
// ---------------------------------------------------------------------------
__global__ __launch_bounds__(256)
void k_step(int t, const float* __restrict__ gate_pre,
            const ushort_t* __restrict__ Wbf,
            const ushort_t* __restrict__ hin, ushort_t* __restrict__ hout,
            float* __restrict__ cx, float* __restrict__ outs) {
    int wid  = (blockIdx.x * blockDim.x + threadIdx.x) >> 5;  // 0..255
    int lane = threadIdx.x & 31;
    int btile = wid >> 4, htile = wid & 15;
    int row0 = btile << 4, col0 = htile << 4;
    int moff = (lane >> 4) << 3;
    int col  = col0 + (lane & 15);

    v8f acc[4];
    #pragma unroll
    for (int g = 0; g < 4; ++g) {
        const float* gp = gate_pre + ((size_t)g * TB + (size_t)t * B_SZ) * H_SZ;
        #pragma unroll
        for (int r = 0; r < 8; ++r)
            acc[g][r] = gp[(size_t)(row0 + moff + r) * H_SZ + col];
    }
    #pragma unroll
    for (int k0 = 0; k0 < H_SZ; k0 += 32) {
        v16bf a = load_a_frag(hin, H_SZ, row0, k0);     // shared across gates
        #pragma unroll
        for (int g = 0; g < 4; ++g) {
            v16bf b = load_b_frag(Wbf + (size_t)g * H_SZ * FANIN, FANIN,
                                  col0, D_SZ + k0);     // h-part of the weights
            acc[g] = wmma_bf16(a, b, acc[g]);
        }
    }
    #pragma unroll
    for (int r = 0; r < 8; ++r) {
        int row = row0 + moff + r;
        float fg = sigmoidf_(cosf(acc[0][r]));
        float ig = sigmoidf_(cosf(acc[1][r]));
        float ug = tanhf(cosf(acc[2][r]));
        float og = sigmoidf_(cosf(acc[3][r]));
        size_t idx = (size_t)row * H_SZ + col;
        float cn = fg * cx[idx] + ig * ug;
        float hn = og * tanhf(cn);
        cx[idx]   = cn;
        outs[(size_t)t * B_SZ * H_SZ + idx] = hn;
        hout[idx] = f2bf(hn);
    }
}

// ---------------------------------------------------------------------------
// Attention
// ---------------------------------------------------------------------------
__global__ void k_logits(const float* __restrict__ outs, const float* __restrict__ phi,
                         float* __restrict__ logits) {
    int wid  = (blockIdx.x * blockDim.x + threadIdx.x) >> 5;  // (t,b) pair
    int lane = threadIdx.x & 31;
    int tt = wid >> 8, bb = wid & (B_SZ - 1);
    const float* p = outs + (size_t)wid * H_SZ;
    float s = 0.0f;
    #pragma unroll
    for (int j = 0; j < 8; ++j) {
        int h = lane + 32 * j;
        s += cosf(p[h] + phi[h]);
    }
    #pragma unroll
    for (int off = 16; off; off >>= 1) s += __shfl_xor(s, off, 32);
    if (lane == 0) logits[(size_t)bb * T_LEN + tt] = s;
}

__global__ __launch_bounds__(256)
void k_softmax_ctx(const float* __restrict__ logits, const float* __restrict__ outs,
                   float* __restrict__ context) {
    __shared__ float red[256];
    __shared__ float w[T_LEN];
    int b = blockIdx.x, tid = threadIdx.x;
    const float* lg = logits + (size_t)b * T_LEN;
    float l0 = lg[tid], l1 = lg[tid + 256];
    red[tid] = fmaxf(l0, l1);
    __syncthreads();
    for (int s = 128; s; s >>= 1) {
        if (tid < s) red[tid] = fmaxf(red[tid], red[tid + s]);
        __syncthreads();
    }
    float mx = red[0];
    __syncthreads();
    float e0 = expf(l0 - mx), e1 = expf(l1 - mx);
    red[tid] = e0 + e1;
    __syncthreads();
    for (int s = 128; s; s >>= 1) {
        if (tid < s) red[tid] += red[tid + s];
        __syncthreads();
    }
    float inv = 1.0f / red[0];
    w[tid] = e0 * inv;
    w[tid + 256] = e1 * inv;
    __syncthreads();
    // context[b][h] = sum_t w[t] * outs[t][b][h] ; h = tid (coalesced)
    const float* op = outs + (size_t)b * H_SZ + tid;
    float acc = 0.0f;
    for (int t = 0; t < T_LEN; ++t) acc += w[t] * op[(size_t)t * B_SZ * H_SZ];
    context[(size_t)b * H_SZ + tid] = acc;
}

__global__ void k_tail(const float* __restrict__ outs, const float* __restrict__ cx,
                       float* __restrict__ out) {
    int idx = blockIdx.x * blockDim.x + threadIdx.x;
    if (idx >= B_SZ * H_SZ) return;
    const size_t TBH = (size_t)T_LEN * B_SZ * H_SZ;
    out[TBH + idx] = outs[(size_t)(T_LEN - 1) * B_SZ * H_SZ + idx];  // hx
    out[TBH + (size_t)B_SZ * H_SZ + idx] = cx[idx];                  // cx
}

__global__ void k_broadcast(const float* __restrict__ context, float* __restrict__ out) {
    size_t idx = (size_t)blockIdx.x * blockDim.x + threadIdx.x;
    if (idx < (size_t)T_LEN * B_SZ * H_SZ)
        out[idx] = context[idx & (size_t)(B_SZ * H_SZ - 1)];
}

// ---------------------------------------------------------------------------
// Host side
// ---------------------------------------------------------------------------
extern "C" void kernel_launch(void* const* d_in, const int* in_sizes, int n_in,
                              void* d_out, int out_size, void* d_ws, size_t ws_size,
                              hipStream_t stream) {
    const float* x   = (const float*)d_in[0];
    const float* Wf  = (const float*)d_in[1];
    const float* bf_ = (const float*)d_in[2];
    const float* Wi  = (const float*)d_in[3];
    const float* bi  = (const float*)d_in[4];
    const float* Wu  = (const float*)d_in[5];
    const float* bu  = (const float*)d_in[6];
    const float* Wo  = (const float*)d_in[7];
    const float* bo  = (const float*)d_in[8];
    const float* thf = (const float*)d_in[9];
    const float* thi = (const float*)d_in[10];
    const float* thu = (const float*)d_in[11];
    const float* tho = (const float*)d_in[12];
    const float* phi = (const float*)d_in[13];

    char* ws = (char*)d_ws;
    size_t off = 0;
    auto take = [&](size_t bytes) { char* p = ws + off; off += (bytes + 255) & ~(size_t)255; return p; };

    ushort_t* Wbf     = (ushort_t*)take((size_t)4 * H_SZ * FANIN * 2);          // 1 MiB
    float*    bias    = (float*)   take((size_t)4 * H_SZ * 4);
    ushort_t* xbf     = (ushort_t*)take((size_t)TB * D_SZ * 2);                 // 64 MiB
    float*    gatepre = (float*)   take((size_t)4 * TB * H_SZ * 4);             // 512 MiB
    ushort_t* hbuf0   = (ushort_t*)take((size_t)B_SZ * H_SZ * 2);
    ushort_t* hbuf1   = (ushort_t*)take((size_t)B_SZ * H_SZ * 2);
    float*    cx      = (float*)   take((size_t)B_SZ * H_SZ * 4);
    float*    logits  = (float*)   take((size_t)B_SZ * T_LEN * 4);
    float*    context = (float*)   take((size_t)B_SZ * H_SZ * 4);

    float* out  = (float*)d_out;
    float* outs = out;  // outs[T][B][H] staged in d_out, overwritten at the end

    // --- conversions / init ---
    {
        int n = 4 * H_SZ * FANIN;
        k_conv_weights<<<(n + 255) / 256, 256, 0, stream>>>(Wf, Wi, Wu, Wo, Wbf);
    }
    k_bias<<<(4 * H_SZ + 255) / 256, 256, 0, stream>>>(bf_, thf, bi, thi, bu, thu, bo, tho, bias);
    {
        int n = TB * D_SZ;
        k_conv_inputs<<<(n + 255) / 256, 256, 0, stream>>>(x, xbf, n);
    }
    k_init_state<<<(B_SZ * H_SZ + 255) / 256, 256, 0, stream>>>(hbuf0, cx);

    // --- hoisted x-part GEMM: 4 * 131072 * 256 output, K=256 ---
    k_xgemm<<<65536, 256, 0, stream>>>(xbf, Wbf, bias, gatepre);

    // --- sequential recurrence: one launch per timestep ---
    for (int t = 0; t < T_LEN; ++t) {
        const ushort_t* hin = (t & 1) ? hbuf1 : hbuf0;
        ushort_t*      hout = (t & 1) ? hbuf0 : hbuf1;
        k_step<<<32, 256, 0, stream>>>(t, gatepre, Wbf, hin, hout, cx, outs);
    }

    // --- attention: logits, softmax over T, context, broadcast ---
    k_logits<<<TB / 8, 256, 0, stream>>>(outs, phi, logits);
    k_softmax_ctx<<<B_SZ, 256, 0, stream>>>(logits, outs, context);
    k_tail<<<(B_SZ * H_SZ + 255) / 256, 256, 0, stream>>>(outs, cx, out);
    k_broadcast<<<(T_LEN * B_SZ * H_SZ + 255) / 256, 256, 0, stream>>>(context, out);
}
